// AdvancedKANLayer_38585986187663
// MI455X (gfx1250) — compile-verified, matrix operand order checked by reading, not run on hardware
//
#include <hip/hip_runtime.h>

typedef __attribute__((ext_vector_type(16))) _Float16 v16h;
typedef __attribute__((ext_vector_type(8)))  float    v8f;

#define B_N   2048
#define I_N   64
#define J_N   64
#define G_N   8
#define K_N   (I_N * G_N)      // 512 contraction length
#define BT    32               // batch rows per workgroup
#define PADK  (K_N + 8)        // +8 halfs (16B) row pad -> rows 4 banks apart
#define KSTEPS (K_N / 32)      // 16 WMMA K-steps

// ---------------------------------------------------------------------------
// Kernel 1: fold scale into spline coeffs, emit f16 weight matrix transposed:
//   Wt[j][i*8+g] = coeffs[i,j,g] * scale[i,j]
// ---------------------------------------------------------------------------
__global__ void kan_prep(const float* __restrict__ coeffs,
                         const float* __restrict__ scale,
                         _Float16* __restrict__ wt) {
  int idx = blockIdx.x * blockDim.x + threadIdx.x;    // 0 .. J*K-1
  if (idx >= J_N * K_N) return;
  int j  = idx >> 9;          // / 512
  int ig = idx & (K_N - 1);
  int i  = ig >> 3;
  int g  = ig & 7;
  float w = coeffs[(i * J_N + j) * G_N + g] * scale[i * J_N + j];
  wt[idx] = (_Float16)w;
}

// ---------------------------------------------------------------------------
// Kernel 2: fused tanh -> RBF basis (LDS, f16) -> WMMA GEMM -> +bias
// 64 threads (2 waves), BT=32 rows per block. Each wave: 16 rows x 64 cols.
// B fragments double-buffered in registers so the 8 global_load_b128 for
// K-step k+1 overlap the 4 v_wmma of K-step k (partial s_wait_loadcnt).
// ---------------------------------------------------------------------------
__global__ __launch_bounds__(64)
void kan_main(const float* __restrict__ x,
              const float* __restrict__ knots,      // k[g] = knots[0,0,g]
              const float* __restrict__ bias,
              const _Float16* __restrict__ wt,      // [J][K] f16
              float* __restrict__ out) {
  __shared__ __align__(16) _Float16 sA[BT][PADK];

  const int tid = threadIdx.x;        // 0..63
  const int b0  = blockIdx.x * BT;

  // Warm L2/WGP$ with the shared weight matrix (global_prefetch_b8).
  __builtin_prefetch(wt + (size_t)tid * K_N, 0, 1);

  // Knot grid is shared across (i,j): broadcast linspace in this problem.
  float k[G_N];
#pragma unroll
  for (int g = 0; g < G_N; ++g) k[g] = knots[g];

  // ---- Phase A: basis[b, i*8+g] = exp(-2*(tanh(x[b,i]) - k[g])^2) ---------
  const int i = tid;                  // thread owns input dim i
  const float L2E = 1.4426950408889634f;   // log2(e)
  for (int r = 0; r < BT; ++r) {
    float xv = x[(size_t)(b0 + r) * I_N + i];   // coalesced across tid
#if __has_builtin(__builtin_amdgcn_tanhf)
    float xn = __builtin_amdgcn_tanhf(xv);
#else
    float xn = tanhf(xv);
#endif
    union { _Float16 h[8]; uint4 u; } pk;
#pragma unroll
    for (int g = 0; g < G_N; ++g) {
      float d  = xn - k[g];
      float e2 = -2.0f * d * d * L2E;
#if __has_builtin(__builtin_amdgcn_exp2f)
      pk.h[g] = (_Float16)__builtin_amdgcn_exp2f(e2);
#else
      pk.h[g] = (_Float16)exp2f(e2);
#endif
    }
    *(uint4*)(&sA[r][i * G_N]) = pk.u;          // 16B LDS store
  }
  __syncthreads();

  // ---- Phase C: out[16x64 strip] = A[16x512] * Wt^T[512x64] ---------------
  const int lane  = tid & 31;
  const int wave  = tid >> 5;         // 0..1
  const int rBase = wave * 16;
  const int mrow  = lane & 15;
  const int sel   = lane >> 4;        // 0: K {0..7,16..23}, 1: K {8..15,24..31}

  union Frag { uint4 u[2]; v16h v; };

  // Per-lane base pointers (compiler folds tile/kstep terms into immediates).
  const _Float16* wtLane = wt + (size_t)mrow * K_N + sel * 8;
  const _Float16* aLane  = &sA[rBase + mrow][sel * 8];

  v8f acc[4];
#pragma unroll
  for (int n = 0; n < 4; ++n)
    acc[n] = (v8f){0.f, 0.f, 0.f, 0.f, 0.f, 0.f, 0.f, 0.f};

  Frag bf[2][4];   // double-buffered B fragments (ping-pong registers)
  Frag af[2];      // double-buffered A fragment

  // Prologue: issue loads for K-step 0.
#pragma unroll
  for (int n = 0; n < 4; ++n) {
    const _Float16* bp = wtLane + (size_t)(n * 16) * K_N;
    bf[0][n].u[0] = *(const uint4*)(bp);
    bf[0][n].u[1] = *(const uint4*)(bp + 16);
  }
  af[0].u[0] = *(const uint4*)(aLane);
  af[0].u[1] = *(const uint4*)(aLane + 16);

#pragma unroll
  for (int ks = 0; ks < KSTEPS; ++ks) {
    const int cur = ks & 1;
    const int nxt = cur ^ 1;

    // Issue next K-step's loads before consuming this step's fragments.
    if (ks + 1 < KSTEPS) {
      const int k1 = (ks + 1) * 32;
#pragma unroll
      for (int n = 0; n < 4; ++n) {
        const _Float16* bp = wtLane + (size_t)(n * 16) * K_N + k1;
        bf[nxt][n].u[0] = *(const uint4*)(bp);
        bf[nxt][n].u[1] = *(const uint4*)(bp + 16);
      }
      af[nxt].u[0] = *(const uint4*)(aLane + k1);
      af[nxt].u[1] = *(const uint4*)(aLane + k1 + 16);
    }

#pragma unroll
    for (int n = 0; n < 4; ++n) {
      acc[n] = __builtin_amdgcn_wmma_f32_16x16x32_f16(
          /*neg_a=*/false, af[cur].v, /*neg_b=*/false, bf[cur][n].v,
          /*c_mod=*/(short)0, acc[n], /*reuse_a=*/false, /*reuse_b=*/false);
    }
  }

  // ---- Epilogue: +bias, store f32 (C/D layout: M = v + 8*sel, N = mrow) ---
#pragma unroll
  for (int n = 0; n < 4; ++n) {
    const int col = n * 16 + mrow;
    const float bv = bias[col];
#pragma unroll
    for (int v = 0; v < 8; ++v) {
      const int row = rBase + v + sel * 8;
      out[(size_t)(b0 + row) * J_N + col] = acc[n][v] + bv;
    }
  }
}

// ---------------------------------------------------------------------------
extern "C" void kernel_launch(void* const* d_in, const int* in_sizes, int n_in,
                              void* d_out, int out_size, void* d_ws, size_t ws_size,
                              hipStream_t stream) {
  const float* x      = (const float*)d_in[0];   // [B, I]
  const float* coeffs = (const float*)d_in[1];   // [I, J, G]
  const float* knots  = (const float*)d_in[2];   // [I, J, G] (broadcast over i,j)
  const float* scale  = (const float*)d_in[3];   // [I, J]
  const float* bias   = (const float*)d_in[4];   // [J]
  float* out          = (float*)d_out;           // [B, J]

  _Float16* wt = (_Float16*)d_ws;                // J*K f16 = 64 KB scratch

  kan_prep<<<(J_N * K_N + 255) / 256, 256, 0, stream>>>(coeffs, scale, wt);
  kan_main<<<B_N / BT, 64, 0, stream>>>(x, knots, bias, wt, out);
}
